// RelationalGraphLayer_70557722739396
// MI455X (gfx1250) — matrix operand order repaired and checked by reading, not compiled, not run on hardware
//
#include <hip/hip_runtime.h>
#include <hip/hip_bf16.h>

typedef __bf16 bf16_t;
typedef __attribute__((ext_vector_type(8)))  __bf16 v8bf;
typedef __attribute__((ext_vector_type(16))) __bf16 v16bf;
typedef __attribute__((ext_vector_type(8)))  float  v8f;
typedef __attribute__((ext_vector_type(4)))  int          v4i;
typedef __attribute__((ext_vector_type(4)))  unsigned int u32x4;
typedef __attribute__((ext_vector_type(8)))  int          i32x8;
typedef __attribute__((ext_vector_type(4)))  int          i32x4;

#define N_NODES 20000
#define N_EDGES 150000
#define DIM 512
#define M_PAD 20096   // 157 * 128, padded row count for GEMM tiles

// ---- CDNA5 data-movement probes (compile-safe: fall back if absent) -------
#if __has_builtin(__builtin_amdgcn_tensor_load_to_lds) && __has_builtin(__builtin_amdgcn_s_wait_tensorcnt)
#define HAVE_TDM 1
#else
#define HAVE_TDM 0
#endif

#if __has_builtin(__builtin_amdgcn_global_load_async_to_lds_b128) && __has_builtin(__builtin_amdgcn_s_wait_asynccnt)
#define HAVE_ASYNC 1
#else
#define HAVE_ASYNC 0
#endif

// async builtin takes typed b128 pointers: (v4i AS1*, v4i AS3*, imm, imm)
#define AS1V4(p) ((__attribute__((address_space(1))) v4i*)(unsigned long long)(const void*)(p))
#define AS3V4(p) ((__attribute__((address_space(3))) v4i*)(void*)(p))

__device__ __forceinline__ unsigned lds_byte_offset(const void* p) {
    return (unsigned)(unsigned long long)
        ( (__attribute__((address_space(3))) char*)(void*)(p) );
}

#if HAVE_TDM
// Issue one TDM 2D tile load: tile_d1 rows x tile_d0 bf16 elements, global row
// stride row_stride elements, LDS destination padded by 16B every 128B row
// (matches the [128][72] bf16 LDS layout: pad_interval=32 DWORDs, pad=4 DWORDs).
__device__ __forceinline__ void tdm_load_tile(unsigned lds_off, const void* gptr,
                                              int tile_d0, int tile_d1,
                                              int row_stride) {
    unsigned long long ga = (unsigned long long)gptr;
    u32x4 g0;
    g0[0] = 1u;                                          // count=1 (valid), user mode
    g0[1] = lds_off;                                     // lds_addr (bytes)
    g0[2] = (unsigned)(ga & 0xffffffffu);                // global_addr[31:0]
    g0[3] = (unsigned)((ga >> 32) & 0x1ffffffu)          // global_addr[56:32]
          | (2u << 30);                                  // type=2 ("image")
    i32x8 g1;
    g1[0] = (1 << 16)                                    // data_size = 2 bytes
          | (1 << 20)                                    // pad_enable
          | (4 << 22)                                    // pad_interval: 32 DWORDs (128B)
          | (3 << 25);                                   // pad_amount:   4 DWORDs (16B)
    g1[1] = (tile_d0 & 0xffff) << 16;                    // tensor_dim0[15:0]
    g1[2] = ((tile_d0 >> 16) & 0xffff)                   // tensor_dim0[31:16]
          | ((tile_d1 & 0xffff) << 16);                  // tensor_dim1[15:0]
    g1[3] = ((tile_d1 >> 16) & 0xffff)                   // tensor_dim1[31:16]
          | ((tile_d0 & 0xffff) << 16);                  // tile_dim0
    g1[4] = (tile_d1 & 0xffff);                          // tile_dim1 (tile_dim2=0)
    g1[5] = row_stride;                                  // tensor_dim0_stride[31:0]
    g1[6] = 0;                                           // stride hi / dim1_stride lo
    g1[7] = 0;
    i32x4 z4 = {0, 0, 0, 0};
#if __clang_major__ >= 23
    i32x8 z8 = {0, 0, 0, 0, 0, 0, 0, 0};
    __builtin_amdgcn_tensor_load_to_lds(g0, g1, z4, z4, z8, 0);
#else
    __builtin_amdgcn_tensor_load_to_lds(g0, g1, z4, z4, 0);
#endif
}
#endif

// ---------------------------------------------------------------------------
// fp32 -> bf16 conversion of node features, with zero padding rows
// ---------------------------------------------------------------------------
__global__ __launch_bounds__(256)
void cvt_x_k(const float* __restrict__ x, bf16_t* __restrict__ xb) {
    long i   = (long)blockIdx.x * 256 + threadIdx.x;   // over M_PAD*DIM
    long row = i >> 9;
    xb[i] = (row < N_NODES) ? (bf16_t)x[i] : (bf16_t)0.0f;
}

// fp32 W[r][k][n] -> bf16 Wt[r][n][k]  (transpose so B fragments are contiguous)
__global__ __launch_bounds__(256)
void cvt_wt_k(const float* __restrict__ W, bf16_t* __restrict__ Wt) {
    long i   = (long)blockIdx.x * 256 + threadIdx.x;   // over 3*DIM*DIM
    long r   = i >> 18;
    long rem = i & 262143;
    long n   = rem >> 9;
    long k   = rem & 511;
    Wt[i] = (bf16_t)W[(r << 18) + (k << 9) + n];
}

// ---------------------------------------------------------------------------
// WMMA GEMM:  C[M,512] = A[M,512](bf16) x W[512,512](bf16, stored transposed)
// Block tile 128x128, 8 waves, each wave 32(M) x 64(N) = 2x4 WMMA fragments.
// A tile staged by the Tensor Data Mover (wave 0), B tile by async global->LDS
// loads; falls back to load+ds_store if the builtins are unavailable.
// SELF=true: predicated store (node_type[row]==blockIdx.z), writes d_out.
// SELF=false: unconditional store of xr + b_rel into workspace.
// ---------------------------------------------------------------------------
template<bool SELF>
__global__ __launch_bounds__(256)
void rgcn_gemm(const bf16_t* __restrict__ A,     // [M_PAD][DIM]
               const bf16_t* __restrict__ Wt,    // [DIM][DIM] n-major (base; SELF indexes by z)
               const float*  __restrict__ bias,  // [DIM] (base; SELF indexes by z)
               const int*    __restrict__ ntype, // [N_NODES] (SELF only)
               float* __restrict__ C)            // [N_NODES][DIM]
{
    __shared__ bf16_t As[128][72];   // 64 data + 8 pad bf16: 144B row stride
    __shared__ bf16_t Bs[128][72];

    const int tid  = threadIdx.x;
    const int wid  = tid >> 5;
    const int lane = tid & 31;
    const int half = lane >> 4;     // 0: lanes 0-15, 1: lanes 16-31
    const int ln   = lane & 15;
    const int wm   = wid & 3;       // wave M group (x32 rows)
    const int wn   = wid >> 2;      // wave N group (x64 cols)

    int sel = 0;
    if (SELF) {
        sel = blockIdx.z;
        Wt   += (long)sel * DIM * DIM;
        bias += sel * DIM;
    }

    const long mbase = (long)blockIdx.y * 128;
    const int  nbase = blockIdx.x * 128;

    v8f acc[2][4];
    #pragma unroll
    for (int i = 0; i < 2; i++)
        #pragma unroll
        for (int j = 0; j < 4; j++)
            #pragma unroll
            for (int q = 0; q < 8; q++) acc[i][j][q] = 0.0f;

    for (int k0 = 0; k0 < DIM; k0 += 64) {
        // ---- stage B tile (128 x 64 bf16), per-lane 16B chunks ------------
        #pragma unroll
        for (int i = 0; i < 4; i++) {
            int chunk = i * 256 + tid;          // 0..1023
            int row   = chunk >> 3;             // 0..127
            int kc    = (chunk & 7) * 8;        // 0..56
            const bf16_t* gB = &Wt[(long)(nbase + row) * DIM + k0 + kc];
#if HAVE_ASYNC
            __builtin_amdgcn_global_load_async_to_lds_b128(
                AS1V4(gB), AS3V4(&Bs[row][kc]), 0, 0);
#else
            *(v8bf*)&Bs[row][kc] = *(const v8bf*)gB;
#endif
        }
        // ---- stage A tile (128 rows x 64 bf16) ----------------------------
#if HAVE_TDM
        if (wid == 0) {
            tdm_load_tile(lds_byte_offset(&As[0][0]),
                          &A[mbase * DIM + k0],
                          /*tile_d0=*/64, /*tile_d1=*/128, /*row_stride=*/DIM);
        }
#else
        #pragma unroll
        for (int i = 0; i < 4; i++) {
            int chunk = i * 256 + tid;
            int row   = chunk >> 3;
            int kc    = (chunk & 7) * 8;
            const bf16_t* gA = &A[(mbase + row) * DIM + k0 + kc];
#if HAVE_ASYNC
            __builtin_amdgcn_global_load_async_to_lds_b128(
                AS1V4(gA), AS3V4(&As[row][kc]), 0, 0);
#else
            *(v8bf*)&As[row][kc] = *(const v8bf*)gA;
#endif
        }
#endif
        // ---- wait for staging, then make visible to all waves -------------
#if HAVE_ASYNC
        __builtin_amdgcn_s_wait_asynccnt(0);
#endif
#if HAVE_TDM
        if (wid == 0) __builtin_amdgcn_s_wait_tensorcnt(0);
#endif
        __syncthreads();

        #pragma unroll
        for (int kk = 0; kk < 64; kk += 32) {
            v16bf af[2];
            v16bf bf[4];
            // A fragment: row = M, lane-half selects K {0..7,16..23} vs {8..15,24..31}
            #pragma unroll
            for (int mf = 0; mf < 2; mf++) {
                int r = wm * 32 + mf * 16 + ln;
                v8bf lo = *(const v8bf*)&As[r][kk + half * 8];
                v8bf hi = *(const v8bf*)&As[r][kk + half * 8 + 16];
                #pragma unroll
                for (int q = 0; q < 8; q++) { af[mf][q] = lo[q]; af[mf][8 + q] = hi[q]; }
            }
            // B fragment: row of Bs = output column N, lane-half selects K {0..7,16..23} vs {8..15,24..31}
            #pragma unroll
            for (int nf = 0; nf < 4; nf++) {
                int r = wn * 64 + nf * 16 + ln;
                v8bf lo = *(const v8bf*)&Bs[r][kk + half * 8];
                v8bf hi = *(const v8bf*)&Bs[r][kk + half * 8 + 16];
                #pragma unroll
                for (int q = 0; q < 8; q++) { bf[nf][q] = lo[q]; bf[nf][8 + q] = hi[q]; }
            }
            #pragma unroll
            for (int mf = 0; mf < 2; mf++)
                #pragma unroll
                for (int nf = 0; nf < 4; nf++)
                    acc[mf][nf] = __builtin_amdgcn_wmma_f32_16x16x32_bf16(
                        false, af[mf], false, bf[nf],
                        (short)0, acc[mf][nf], false, false);
        }
        __syncthreads();
    }

    // epilogue: C layout — VGPR j holds rows (base+j) [lanes 0-15] / (base+j+8) [16-31]
    #pragma unroll
    for (int mf = 0; mf < 2; mf++) {
        #pragma unroll
        for (int nf = 0; nf < 4; nf++) {
            int   col = nbase + wn * 64 + nf * 16 + ln;
            float b   = bias[col];
            #pragma unroll
            for (int j = 0; j < 8; j++) {
                long row = mbase + wm * 32 + mf * 16 + half * 8 + j;
                if (row < N_NODES) {
                    if (!SELF || ntype[row] == sel)
                        C[row * DIM + col] = acc[mf][nf][j] + b;
                }
            }
        }
    }
}

// ---------------------------------------------------------------------------
// Edge scatter: one block per edge; bias was folded into xr in the GEMM.
// 512 f32 atomic adds per edge stay resident in the 192MB L2 (out = 41MB).
// ---------------------------------------------------------------------------
__global__ __launch_bounds__(256)
void scatter_edges(const float* __restrict__ xr,
                   const int* __restrict__ esrc,
                   const int* __restrict__ edst,
                   const int* __restrict__ etype,
                   int rel,
                   float* __restrict__ out)
{
    int e = blockIdx.x;
    if (etype[e] != rel) return;
    long s = esrc[e];
    long d = edst[e];
    const float* __restrict__ srcp = xr + s * DIM;
    float* __restrict__ dstp = out + d * DIM;
    #pragma unroll
    for (int c = threadIdx.x; c < DIM; c += 256)
        atomicAdd(&dstp[c], srcp[c]);
}

__global__ __launch_bounds__(256)
void relu_k(float* __restrict__ out, int n) {
    int i = blockIdx.x * 256 + threadIdx.x;
    if (i < n) out[i] = fmaxf(out[i], 0.0f);
}

// ---------------------------------------------------------------------------
extern "C" void kernel_launch(void* const* d_in, const int* in_sizes, int n_in,
                              void* d_out, int out_size, void* d_ws, size_t ws_size,
                              hipStream_t stream)
{
    const float* x      = (const float*)d_in[0];
    const float* W_rel  = (const float*)d_in[1];
    const float* b_rel  = (const float*)d_in[2];
    const float* W_self = (const float*)d_in[3];
    const float* b_self = (const float*)d_in[4];
    const int*   esrc   = (const int*)d_in[5];
    const int*   edst   = (const int*)d_in[6];
    const int*   etype  = (const int*)d_in[7];
    const int*   ntype  = (const int*)d_in[8];
    float* out = (float*)d_out;

    // workspace layout (~65 MB): bf16 x (padded), bf16 Wt_rel, bf16 Wt_self, fp32 xr
    char* ws = (char*)d_ws;
    bf16_t* xb  = (bf16_t*)ws;  ws += (size_t)M_PAD * DIM * sizeof(bf16_t);
    bf16_t* WtR = (bf16_t*)ws;  ws += (size_t)3 * DIM * DIM * sizeof(bf16_t);
    bf16_t* WtS = (bf16_t*)ws;  ws += (size_t)3 * DIM * DIM * sizeof(bf16_t);
    float*  xr  = (float*)ws;   // [N_NODES][DIM]

    cvt_x_k <<<M_PAD * DIM / 256,  256, 0, stream>>>(x, xb);
    cvt_wt_k<<<3 * DIM * DIM / 256, 256, 0, stream>>>(W_rel,  WtR);
    cvt_wt_k<<<3 * DIM * DIM / 256, 256, 0, stream>>>(W_self, WtS);

    // self transform: z = node type, predicated store initializes every out row
    rgcn_gemm<true><<<dim3(DIM / 128, M_PAD / 128, 3), 256, 0, stream>>>(
        xb, WtS, b_self, ntype, out);

    // per relation: dense transform into xr (bias folded in), then scatter-add
    for (int r = 0; r < 3; r++) {
        rgcn_gemm<false><<<dim3(DIM / 128, M_PAD / 128, 1), 256, 0, stream>>>(
            xb, WtR + (size_t)r * DIM * DIM, b_rel + r * DIM, nullptr, xr);
        scatter_edges<<<N_EDGES, 256, 0, stream>>>(xr, esrc, edst, etype, r, out);
    }

    relu_k<<<(N_NODES * DIM + 255) / 256, 256, 0, stream>>>(out, N_NODES * DIM);
}
